// TriangleMultiplication_7026566497088
// MI455X (gfx1250) — matrix operand order repaired and checked by reading, not compiled, not run on hardware
//
#include <hip/hip_runtime.h>
#include <hip/hip_bf16.h>

// ---------------------------------------------------------------------------
// TriangleMultiplication (incoming) for MI455X / gfx1250
// B=2, N=256, D=256  -> R = B*N*N = 131072 rows of width 256 everywhere.
// All GEMMs run on v_wmma_f32_16x16x32_bf16 (bf16 A/B, f32 accumulate).
//
// v2: a/b are stored pre-transposed per (b,n) slice by the in-projection
//     epilogue (their only consumer is the tri GEMM), so every fragment load
//     in every GEMM is a contiguous b128 load -- no address gathers anywhere.
//     Weights are converted f32->bf16 once up front instead of per-fragment.
// ---------------------------------------------------------------------------

typedef __bf16 bf16_t;
typedef __attribute__((ext_vector_type(16))) __bf16 v16bf;
typedef __attribute__((ext_vector_type(8)))  __bf16 v8bf;
typedef __attribute__((ext_vector_type(4)))  __bf16 v4bf;
typedef __attribute__((ext_vector_type(8)))  float  v8f;

#define LN_EPS 1e-5f

static constexpr int D_DIM = 256;
static constexpr int R_ROWS = 2 * 256 * 256;        // 131072
static constexpr int SLICE = D_DIM * D_DIM;         // 65536 elems per (b,n)

// ---------------------------------------------------------------------------
// WMMA wrapper: D = A(16x32 bf16) * B(32x16 bf16) + C(16x16 f32)
// ---------------------------------------------------------------------------
__device__ __forceinline__ v8f wmma_bf16(v16bf a, v16bf b, v8f c) {
  return __builtin_amdgcn_wmma_f32_16x16x32_bf16(
      /*neg_a=*/false, a, /*neg_b=*/false, b,
      /*c_mod=*/(short)0, c, /*reuse_a=*/false, /*reuse_b=*/false);
}

// A fragment from row-major X[m][k] (ld = row stride, elements).
// ISA 7.12.2 (16-bit A 16x32): lane = M mod 16; lane-half: e=0..7 -> K=8h+e,
// e=8..15 -> K=8h+8+e. Two 16B loads per lane.
__device__ __forceinline__ v16bf load_a_rm(const bf16_t* __restrict__ p, int ld,
                                           int m0, int k0) {
  const int lane = threadIdx.x & 31;
  const bf16_t* base =
      p + (size_t)(m0 + (lane & 15)) * ld + k0 + ((lane >> 4) << 3);
  v8bf lo = *(const v8bf*)base;          // K: +0..+7
  v8bf hi = *(const v8bf*)(base + 16);   // K: +16..+23
  return __builtin_shufflevector(lo, hi, 0, 1, 2, 3, 4, 5, 6, 7,
                                 8, 9, 10, 11, 12, 13, 14, 15);
}

// B fragment where logical B[k][n] is stored n-major: p[n][k] (ld = k stride).
// ISA layout (dense 32x16 bf16 B): lane = N mod 16; e=0..7 -> K=8h+e,
// e=8..15 -> K=16+8h+(e-8) (matches A). Two 16B loads per lane.
// Also used for Y = X*W^T with bf16 W[n][k] row-major (identical shape).
__device__ __forceinline__ v16bf load_b_nmajor(const bf16_t* __restrict__ p,
                                               int ld, int n0, int k0) {
  const int lane = threadIdx.x & 31;
  const bf16_t* base =
      p + (size_t)(n0 + (lane & 15)) * ld + k0 + ((lane >> 4) << 3);
  v8bf lo = *(const v8bf*)base;          // K: +0..+7
  v8bf hi = *(const v8bf*)(base + 16);   // K: +16..+23
  return __builtin_shufflevector(lo, hi, 0, 1, 2, 3, 4, 5, 6, 7,
                                 8, 9, 10, 11, 12, 13, 14, 15);
}

__device__ __forceinline__ float sigmoid_f(float v) {
  return 1.0f / (1.0f + __expf(-v));
}

// ---------------------------------------------------------------------------
// Kernel 0: concatenate + convert weights f32 -> bf16 once.
// wcat rows: [0,512) = w_in, [512,768) = w_gate, [768,1024) = w_out.
// 1024*256 = 262144 elems; 256 blocks x 256 threads x 4 elems.
// ---------------------------------------------------------------------------
__global__ __launch_bounds__(256) void cvt_weights(
    const float* __restrict__ w_in, const float* __restrict__ w_gate,
    const float* __restrict__ w_out, bf16_t* __restrict__ wcat) {
  const size_t i = ((size_t)blockIdx.x * 256 + threadIdx.x) * 4;
  float4 v;
  if (i < 131072)          v = *(const float4*)(w_in + i);
  else if (i < 196608)     v = *(const float4*)(w_gate + (i - 131072));
  else                     v = *(const float4*)(w_out + (i - 196608));
  v4bf o;
  o[0] = (bf16_t)v.x; o[1] = (bf16_t)v.y; o[2] = (bf16_t)v.z; o[3] = (bf16_t)v.w;
  *(v4bf*)(wcat + i) = o;
}

// ---------------------------------------------------------------------------
// Kernel 1/4: row LayerNorm over 256 f32 -> bf16. One wave per row.
// ---------------------------------------------------------------------------
__global__ __launch_bounds__(256) void ln_rows_bf16(
    const float* __restrict__ in, const float* __restrict__ gamma,
    const float* __restrict__ beta, bf16_t* __restrict__ out) {
  const int wave = threadIdx.x >> 5;
  const int lane = threadIdx.x & 31;
  const size_t row = (size_t)blockIdx.x * 8 + wave;
  const float* p = in + row * D_DIM;

  float4 v0 = ((const float4*)p)[lane * 2 + 0];
  float4 v1 = ((const float4*)p)[lane * 2 + 1];
  float x[8] = {v0.x, v0.y, v0.z, v0.w, v1.x, v1.y, v1.z, v1.w};

  float s = 0.f, ss = 0.f;
#pragma unroll
  for (int i = 0; i < 8; ++i) { s += x[i]; ss += x[i] * x[i]; }
#pragma unroll
  for (int off = 16; off > 0; off >>= 1) {
    s  += __shfl_xor(s, off, 32);
    ss += __shfl_xor(ss, off, 32);
  }
  const float mean = s * (1.0f / D_DIM);
  const float var  = ss * (1.0f / D_DIM) - mean * mean;
  const float rstd = rsqrtf(var + LN_EPS);

  bf16_t* q = out + row * D_DIM;
  const int c0 = lane * 8;
#pragma unroll
  for (int i = 0; i < 8; ++i) {
    const int c = c0 + i;
    q[c] = (bf16_t)((x[i] - mean) * rstd * gamma[c] + beta[c]);
  }
}

// ---------------------------------------------------------------------------
// Kernel 2: fused in/gate projection, Y = x_ln @ wcat[0:768]^T.
// cols 0..255   -> a = SiLU(.)   stored TRANSPOSED: a_t[bn][i][k]   (bf16)
// cols 256..511 -> b = SiLU(.)   stored TRANSPOSED: b_t[bn][j][k]   (bf16)
// cols 512..767 -> g = sigmoid(.) stored row-major g[row][e]        (bf16)
// Transposed store is one packed 16B v8bf per tile: the 8 accum values per
// lane run contiguously along k (rows = (bn,k), tiles 16-aligned in k).
// Wave tile: 16 rows x 64 cols; block = 8 waves = 128 rows x 64 cols.
// ---------------------------------------------------------------------------
__global__ __launch_bounds__(256) void proj_in_gate(
    const bf16_t* __restrict__ xln, const bf16_t* __restrict__ wcat,
    bf16_t* __restrict__ a_t, bf16_t* __restrict__ b_t,
    bf16_t* __restrict__ g_out) {
  const int wave = threadIdx.x >> 5;
  const int lane = threadIdx.x & 31;
  const int row0 = blockIdx.x * 128 + wave * 16;   // flattened (bn, k)
  const int col0 = blockIdx.y * 64;                // 0..704

  v8f acc[4] = {};
  for (int k0 = 0; k0 < D_DIM; k0 += 32) {
    v16bf af = load_a_rm(xln, D_DIM, row0, k0);
    if (k0 + 32 < D_DIM)
      __builtin_prefetch(xln + (size_t)(row0 + (lane & 15)) * D_DIM + k0 + 32,
                         0, 1);
#pragma unroll
    for (int t = 0; t < 4; ++t) {
      v16bf bf = load_b_nmajor(wcat, D_DIM, col0 + 16 * t, k0);
      acc[t] = wmma_bf16(af, bf, acc[t]);
    }
  }

  const int n = lane & 15, half = lane >> 4;
  const int bn = row0 >> 8;                 // (b*N + n) slice index
  const int kbase = (row0 & 255) + 8 * half;  // k of acc element r=0

  if (col0 < 512) {
    // a / b: SiLU, packed transposed store (one 16B store per tile).
    bf16_t* dst = (col0 < 256) ? a_t : b_t;
    const int cbias = (col0 < 256) ? 0 : 256;
#pragma unroll
    for (int t = 0; t < 4; ++t) {
      const int col = col0 + 16 * t + n - cbias;   // channel i or j
      v8bf pk;
#pragma unroll
      for (int r = 0; r < 8; ++r) {
        const float v = acc[t][r];
        pk[r] = (bf16_t)(v * sigmoid_f(v));
      }
      *(v8bf*)(dst + (size_t)bn * SLICE + (size_t)col * D_DIM + kbase) = pk;
    }
  } else {
    // gate: sigmoid, row-major store.
#pragma unroll
    for (int t = 0; t < 4; ++t) {
      const int col = col0 + 16 * t + n - 512;
#pragma unroll
      for (int r = 0; r < 8; ++r) {
        const size_t row = (size_t)(row0 + r + 8 * half);
        g_out[row * D_DIM + col] = (bf16_t)sigmoid_f(acc[t][r]);
      }
    }
  }
}

// ---------------------------------------------------------------------------
// Kernel 3: tri[i][j] = sum_k a[k][i] * b[k][j] per (b,n); 512 batches of
// 256^3. Inputs are pre-transposed (a_t[i][k], b_t[j][k]) so both fragment
// loads are contiguous b128s. grid = (8 tiles, 512 batches).
// ---------------------------------------------------------------------------
__global__ __launch_bounds__(256) void tri_outer(
    const bf16_t* __restrict__ a_t, const bf16_t* __restrict__ b_t,
    float* __restrict__ tri) {
  const int wave = threadIdx.x >> 5;
  const int lane = threadIdx.x & 31;
  const int batch = blockIdx.y;
  const int rowblk = blockIdx.x & 1;      // 2 x 128 rows (i)
  const int colblk = blockIdx.x >> 1;     // 4 x 64 cols  (j)

  const bf16_t* as = a_t + (size_t)batch * SLICE;   // [i][k]
  const bf16_t* bs = b_t + (size_t)batch * SLICE;   // [j][k]
  float* ts = tri + (size_t)batch * SLICE;          // [i][j]

  const int row0 = rowblk * 128 + wave * 16;
  const int col0 = colblk * 64;

  v8f acc[4] = {};
  for (int k0 = 0; k0 < D_DIM; k0 += 32) {
    v16bf af = load_a_rm(as, D_DIM, row0, k0);      // A[i][k] = a_t[i][k]
    if (k0 + 32 < D_DIM)
      __builtin_prefetch(as + (size_t)(row0 + (lane & 15)) * D_DIM + k0 + 32,
                         0, 1);
#pragma unroll
    for (int t = 0; t < 4; ++t) {
      v16bf bf = load_b_nmajor(bs, D_DIM, col0 + 16 * t, k0);  // B[k][j]
      acc[t] = wmma_bf16(af, bf, acc[t]);
    }
  }

  const int n = lane & 15, half = lane >> 4;
#pragma unroll
  for (int t = 0; t < 4; ++t)
#pragma unroll
    for (int r = 0; r < 8; ++r)
      ts[(size_t)(row0 + r + 8 * half) * D_DIM + col0 + 16 * t + n] = acc[t][r];
}

// ---------------------------------------------------------------------------
// Kernel 5: out = (LN(tri) @ w_out^T) * g   (f32 output)
// ---------------------------------------------------------------------------
__global__ __launch_bounds__(256) void proj_out_gated(
    const bf16_t* __restrict__ triln, const bf16_t* __restrict__ w_out_bf,
    const bf16_t* __restrict__ g_in, float* __restrict__ out) {
  const int wave = threadIdx.x >> 5;
  const int lane = threadIdx.x & 31;
  const int row0 = blockIdx.x * 128 + wave * 16;
  const int col0 = blockIdx.y * 64;

  v8f acc[4] = {};
  for (int k0 = 0; k0 < D_DIM; k0 += 32) {
    v16bf af = load_a_rm(triln, D_DIM, row0, k0);
    if (k0 + 32 < D_DIM)
      __builtin_prefetch(triln + (size_t)(row0 + (lane & 15)) * D_DIM + k0 + 32,
                         0, 1);
#pragma unroll
    for (int t = 0; t < 4; ++t) {
      v16bf bf = load_b_nmajor(w_out_bf, D_DIM, col0 + 16 * t, k0);
      acc[t] = wmma_bf16(af, bf, acc[t]);
    }
  }

  const int n = lane & 15, half = lane >> 4;
#pragma unroll
  for (int t = 0; t < 4; ++t) {
    const int col = col0 + 16 * t + n;
#pragma unroll
    for (int r = 0; r < 8; ++r) {
      const size_t row = (size_t)(row0 + r + 8 * half);
      out[row * D_DIM + col] = acc[t][r] * (float)g_in[row * D_DIM + col];
    }
  }
}

// ---------------------------------------------------------------------------
// Launch: 6 kernels on `stream`; scratch carved out of d_ws (~403.2 MB).
// ---------------------------------------------------------------------------
extern "C" void kernel_launch(void* const* d_in, const int* in_sizes, int n_in,
                              void* d_out, int out_size, void* d_ws,
                              size_t ws_size, hipStream_t stream) {
  (void)in_sizes; (void)n_in; (void)out_size; (void)ws_size;

  const float* pair   = (const float*)d_in[0];  // [B,N,N,D] f32
  const float* w_in   = (const float*)d_in[1];  // [2D, D]   f32
  const float* w_gate = (const float*)d_in[2];  // [D, D]    f32
  const float* w_out  = (const float*)d_in[3];  // [D, D]    f32
  const float* gamma  = (const float*)d_in[4];  // [D]       f32
  const float* beta   = (const float*)d_in[5];  // [D]       f32
  float* out = (float*)d_out;                   // [B,N,N,D] f32

  char* ws = (char*)d_ws;
  const size_t BF  = (size_t)R_ROWS * D_DIM * sizeof(bf16_t);  // 64 MiB
  const size_t TRI = (size_t)R_ROWS * D_DIM * sizeof(float);   // 128 MiB
  bf16_t* xln  = (bf16_t*)(ws);                 // x_ln bf16 (reused: tri_ln)
  bf16_t* a_t  = (bf16_t*)(ws + 1 * BF);        // a transposed [bn][i][k]
  bf16_t* b_t  = (bf16_t*)(ws + 2 * BF);        // b transposed [bn][j][k]
  bf16_t* g_b  = (bf16_t*)(ws + 3 * BF);        // g row-major
  float*  tri  = (float*) (ws + 4 * BF);        // tri f32
  bf16_t* wcat = (bf16_t*)(ws + 4 * BF + TRI);  // 1024x256 bf16 weights
  bf16_t* triln = xln;                          // x_ln dead after proj_in_gate

  // 0) weights f32 -> bf16, concatenated [w_in; w_gate; w_out].
  cvt_weights<<<dim3(256), 256, 0, stream>>>(w_in, w_gate, w_out, wcat);

  // 1) LN(pair) -> x_ln bf16.
  ln_rows_bf16<<<dim3(R_ROWS / 8), 256, 0, stream>>>(pair, gamma, beta, xln);

  // 2) a_t/b_t = SiLU(x_ln W_in^T) transposed, g = sigmoid(x_ln W_gate^T).
  proj_in_gate<<<dim3(R_ROWS / 128, 768 / 64), 256, 0, stream>>>(
      xln, wcat, a_t, b_t, g_b);

  // 3) tri = a^T b per (b,n): 512 batched 256x256x256 GEMMs, all-b128 loads.
  tri_outer<<<dim3(8, 512), 256, 0, stream>>>(a_t, b_t, tri);

  // 4) LN(tri) -> tri_ln bf16 (reuses x_ln buffer).
  ln_rows_bf16<<<dim3(R_ROWS / 8), 256, 0, stream>>>(tri, gamma, beta, triln);

  // 5) out = (tri_ln W_out^T) * g, f32.
  proj_out_gated<<<dim3(R_ROWS / 128, 256 / 64), 256, 0, stream>>>(
      triln, wcat + 768 * 256, g_b, out);
}
